// MGQA_85555748536828
// MI455X (gfx1250) — matrix operand order repaired and checked by reading, not compiled, not run on hardware
//
#include <hip/hip_runtime.h>
#include <math.h>

// ---------------------------------------------------------------------------
// MI455X (gfx1250) GQA attention block, bf16 WMMA + fp32 softmax.
//   B=4 N=2048 D=2048 H=16 KVH=4 DH=128
// GEMMs use GLOBAL_LOAD_ASYNC_TO_LDS_B128 + double-buffered LDS (ASYNCcnt).
// ---------------------------------------------------------------------------

typedef __attribute__((ext_vector_type(16))) __bf16 v16bf;
typedef __attribute__((ext_vector_type(8)))  __bf16 v8bf;
typedef __attribute__((ext_vector_type(4)))  __bf16 v4bf;
typedef __attribute__((ext_vector_type(8)))  float  v8f;

union BF16x16 { v16bf v; v8bf h[2]; };

__device__ __forceinline__ v8f v8f_zero() {
  v8f z = {0.f, 0.f, 0.f, 0.f, 0.f, 0.f, 0.f, 0.f};
  return z;
}

__device__ __forceinline__ __bf16 f2bf(float f) {
  unsigned u = __builtin_bit_cast(unsigned, f);
  u += 0x7FFFu + ((u >> 16) & 1u);            // round-to-nearest-even
  unsigned short s = (unsigned short)(u >> 16);
  return __builtin_bit_cast(__bf16, s);
}

// Async global -> LDS, 16B per lane, tracked by ASYNCcnt (no VGPR round-trip).
// LDS operand = low 32 bits of the generic pointer (== AS3 byte offset).
__device__ __forceinline__ void async_b128(void* lds, const __bf16* g) {
  asm volatile("global_load_async_to_lds_b128 %0, %1, off"
               :: "v"((unsigned)(unsigned long long)lds), "v"(g)
               : "memory");
}
__device__ __forceinline__ void wait_async0() {
  asm volatile("s_wait_asynccnt 0x0" ::: "memory");
}

// ------------------------- elementwise converts ----------------------------

// fp32 -> bf16, 4 elements / thread
__global__ __launch_bounds__(256) void cvt_f32_bf16(const float* __restrict__ in,
                                                    __bf16* __restrict__ out) {
  int i = (blockIdx.x * 256 + threadIdx.x) * 4;
  float4 f = *(const float4*)(in + i);
  v4bf o;
  o[0] = f2bf(f.x); o[1] = f2bf(f.y); o[2] = f2bf(f.z); o[3] = f2bf(f.w);
  *(v4bf*)(out + i) = o;
}

// W[K][Nout] fp32 -> WT[Nout][K] bf16  (pre-transpose so GEMM B-frags are contiguous)
__global__ __launch_bounds__(256) void transpose_cvt(const float* __restrict__ W,
                                                     __bf16* __restrict__ WT,
                                                     int K, int Nout) {
  int idx = blockIdx.x * 256 + threadIdx.x;
  int k = idx / Nout;
  int n = idx - k * Nout;
  WT[n * K + k] = f2bf(W[idx]);
}

// v slice of qkv[(b*N+n)*3072 + 2560 + kvh*128 + dh]  ->  vt[((b*4+kvh)*128+dh)*N + n]
__global__ __launch_bounds__(256) void transpose_v(const __bf16* __restrict__ qkv,
                                                   __bf16* __restrict__ vt) {
  int idx = blockIdx.x * 256 + threadIdx.x;         // over 4*2048*512
  int dhkv = idx & 511;
  int n    = (idx >> 9) & 2047;
  int b    = idx >> 20;
  int kvh  = dhkv >> 7;
  int dh   = dhkv & 127;
  vt[((b * 4 + kvh) * 128 + dh) * 2048 + n] =
      qkv[(b * 2048 + n) * 3072 + 2560 + dhkv];
}

// ------------------------------ WMMA GEMM ----------------------------------
// C[M x Nout] = A[M x K] * BT[Nout x K]^T, bf16 in, f32 acc.
// Block tile 128x64, 8 waves (4x2), each wave 32x32 (2x2 WMMA tiles), K step 32.
// Double-buffered LDS filled by async-to-LDS copies; fetch of tile k+1 overlaps
// the WMMAs of tile k.
__global__ __launch_bounds__(256) void gemm_bf16(const __bf16* __restrict__ A,
                                                 const __bf16* __restrict__ BT,
                                                 void* __restrict__ Cout,
                                                 int M, int Nout, int K,
                                                 int c_is_f32) {
  __shared__ __align__(16) __bf16 sA[2][128][40];   // 32 + 8 pad (80B row stride)
  __shared__ __align__(16) __bf16 sB[2][64][40];

  const int tid  = threadIdx.x;
  const int lane = tid & 31;
  const int w    = tid >> 5;
  const int wr   = w >> 1;         // 0..3
  const int wc   = w & 1;          // 0..1
  const int mblk = blockIdx.y * 128;
  const int nblk = blockIdx.x * 64;
  const int lrow = lane & 15;
  const int hi    = lane >> 4;     // 0 / 1 half-wave
  const int koff  = hi * 8;        // A fragment K base  (ISA 7.12.2 16-bit A)
  const int kbase = hi * 16;       // unused; B fragment also reads at koff

  // staging coordinates: each thread moves 3 x 16B per K step
  const int rs = tid >> 2;                 // 0..63
  const int cs = (tid & 3) * 8;            // 0,8,16,24
  const __bf16* gA0 = A + (mblk + rs) * K + cs;          // rows 0..63
  const __bf16* gA1 = A + (mblk + 64 + rs) * K + cs;     // rows 64..127
  const __bf16* gB  = BT + (nblk + rs) * K + cs;

  v8f acc[2][2];
  acc[0][0] = v8f_zero(); acc[0][1] = v8f_zero();
  acc[1][0] = v8f_zero(); acc[1][1] = v8f_zero();

  // prologue: fill buffer 0
  async_b128(&sA[0][rs][cs],      gA0);
  async_b128(&sA[0][64 + rs][cs], gA1);
  async_b128(&sB[0][rs][cs],      gB);
  wait_async0();
  __syncthreads();

  for (int k0 = 0; k0 < K; k0 += 32) {
    const int buf = (k0 >> 5) & 1;
    const bool more = (k0 + 32) < K;
    if (more) {                       // prefetch next tile into other buffer
      async_b128(&sA[buf ^ 1][rs][cs],      gA0 + k0 + 32);
      async_b128(&sA[buf ^ 1][64 + rs][cs], gA1 + k0 + 32);
      async_b128(&sB[buf ^ 1][rs][cs],      gB + k0 + 32);
    }

    // ---- build fragments from current buffer and issue WMMA ---------------
    BF16x16 af[2], bfr[2];
#pragma unroll
    for (int rt = 0; rt < 2; ++rt) {
      int m = wr * 32 + rt * 16 + lrow;
      af[rt].h[0] = *(const v8bf*)&sA[buf][m][koff];
      af[rt].h[1] = *(const v8bf*)&sA[buf][m][16 + koff];
    }
#pragma unroll
    for (int nt = 0; nt < 2; ++nt) {
      int n = wc * 32 + nt * 16 + lrow;
      bfr[nt].h[0] = *(const v8bf*)&sB[buf][n][koff];
      bfr[nt].h[1] = *(const v8bf*)&sB[buf][n][16 + koff];
    }
#pragma unroll
    for (int rt = 0; rt < 2; ++rt)
#pragma unroll
      for (int nt = 0; nt < 2; ++nt)
        acc[rt][nt] = __builtin_amdgcn_wmma_f32_16x16x32_bf16(
            false, af[rt].v, false, bfr[nt].v, (short)0, acc[rt][nt],
            false, false);

    if (more) wait_async0();          // own async copies landed
    __syncthreads();                  // everyone's copies visible / reads done
  }

  // ---- epilogue: C layout = N in lanes, M = slot + 8*half -----------------
#pragma unroll
  for (int rt = 0; rt < 2; ++rt)
#pragma unroll
    for (int nt = 0; nt < 2; ++nt)
#pragma unroll
      for (int s = 0; s < 8; ++s) {
        int gm = mblk + wr * 32 + rt * 16 + s + hi * 8;
        int gn = nblk + wc * 32 + nt * 16 + lrow;
        if (c_is_f32)
          ((float*)Cout)[gm * Nout + gn] = acc[rt][nt][s];
        else
          ((__bf16*)Cout)[gm * Nout + gn] = f2bf(acc[rt][nt][s]);
      }
}

// --------------------------- flash attention -------------------------------
// One wave = 16 query rows of one head. grid = (N/64, B*H), block = 128 (4 waves).
// qkv: [b][n][3072]  (q: 0..2047, k: 2048..2559, v: 2560..3071)
// vt : [b][kvh][128][N]   att: [b][n][2048] bf16
__global__ __launch_bounds__(128) void flash_attn(const __bf16* __restrict__ qkv,
                                                  const __bf16* __restrict__ vt,
                                                  __bf16* __restrict__ att) {
  __shared__ __align__(16) __bf16 sP[4][16][32];   // per-wave P tile

  const int tid  = threadIdx.x;
  const int lane = tid & 31;
  const int w    = tid >> 5;
  const int bh   = blockIdx.y;
  const int b    = bh >> 4;        // / H
  const int h    = bh & 15;
  const int kvh  = h & 3;          // GQA: tile(k, r) with r outermost -> h % KVH
  const int qtile = blockIdx.x * 64 + w * 16;
  const int lrow  = lane & 15;
  const int hi    = lane >> 4;
  const int koff  = hi * 8;
  const int kbase = hi * 16;
  const float scale = 0.08838834764831845f;        // 128^-0.5
  const float NEGF  = -3.402823466e38f;

  const __bf16* qb  = qkv + (b * 2048 + qtile) * 3072 + h * 128;
  const __bf16* kb  = qkv + (b * 2048) * 3072 + 2048 + kvh * 128;
  const __bf16* vtb = vt + ((b * 4 + kvh) * 128) * 2048;

  // Q fragments: 16x128 as 4 A-fragments of K=32
  BF16x16 aq[4];
#pragma unroll
  for (int c = 0; c < 4; ++c) {
    const __bf16* p = qb + lrow * 3072 + c * 32;
    aq[c].h[0] = *(const v8bf*)(p + koff);
    aq[c].h[1] = *(const v8bf*)(p + 16 + koff);
  }

  float mrow[8], lsum[8];
  v8f o[8];
#pragma unroll
  for (int s = 0; s < 8; ++s) { mrow[s] = NEGF; lsum[s] = 0.f; }
#pragma unroll
  for (int d = 0; d < 8; ++d) o[d] = v8f_zero();

  for (int jb = 0; jb < qtile + 16; jb += 32) {
    // ---- S = Q K^T for 32 kv columns (two 16x16 tiles, K chained over DH) --
    v8f s0 = v8f_zero(), s1 = v8f_zero();
#pragma unroll
    for (int c = 0; c < 4; ++c) {
      BF16x16 bk0, bk1;
      const __bf16* kp0 = kb + (jb + lrow) * 3072 + c * 32 + koff;
      const __bf16* kp1 = kb + (jb + 16 + lrow) * 3072 + c * 32 + koff;
      bk0.h[0] = *(const v8bf*)kp0;       bk0.h[1] = *(const v8bf*)(kp0 + 16);
      bk1.h[0] = *(const v8bf*)kp1;       bk1.h[1] = *(const v8bf*)(kp1 + 16);
      s0 = __builtin_amdgcn_wmma_f32_16x16x32_bf16(false, aq[c].v, false, bk0.v,
                                                   (short)0, s0, false, false);
      s1 = __builtin_amdgcn_wmma_f32_16x16x32_bf16(false, aq[c].v, false, bk1.v,
                                                   (short)0, s1, false, false);
    }

    // ---- causal mask + online softmax (row lives across 16 lanes) ---------
    float p0[8], p1[8], rcorr[8];
#pragma unroll
    for (int s = 0; s < 8; ++s) {
      int row = qtile + s + hi * 8;
      int c0 = jb + lrow, c1 = c0 + 16;
      float x0 = (c0 <= row) ? s0[s] * scale : NEGF;
      float x1 = (c1 <= row) ? s1[s] * scale : NEGF;
      float t = fmaxf(x0, x1);
      t = fmaxf(t, __shfl_xor(t, 1, 32));
      t = fmaxf(t, __shfl_xor(t, 2, 32));
      t = fmaxf(t, __shfl_xor(t, 4, 32));
      t = fmaxf(t, __shfl_xor(t, 8, 32));
      float mnew = fmaxf(mrow[s], t);
      float corr = __expf(mrow[s] - mnew);
      float e0 = (x0 > -1e38f) ? __expf(x0 - mnew) : 0.f;
      float e1 = (x1 > -1e38f) ? __expf(x1 - mnew) : 0.f;
      float rs2 = e0 + e1;
      rs2 += __shfl_xor(rs2, 1, 32);
      rs2 += __shfl_xor(rs2, 2, 32);
      rs2 += __shfl_xor(rs2, 4, 32);
      rs2 += __shfl_xor(rs2, 8, 32);
      lsum[s] = lsum[s] * corr + rs2;
      mrow[s] = mnew;
      p0[s] = e0; p1[s] = e1; rcorr[s] = corr;
    }
#pragma unroll
    for (int d = 0; d < 8; ++d)
#pragma unroll
      for (int s = 0; s < 8; ++s) o[d][s] *= rcorr[s];

    // ---- P (C layout) -> LDS -> A fragment; same-wave DS ops are in-order --
#pragma unroll
    for (int s = 0; s < 8; ++s) {
      sP[w][s + hi * 8][lrow]      = f2bf(p0[s]);
      sP[w][s + hi * 8][16 + lrow] = f2bf(p1[s]);
    }
    BF16x16 ap;
    ap.h[0] = *(const v8bf*)&sP[w][lrow][koff];
    ap.h[1] = *(const v8bf*)&sP[w][lrow][16 + koff];

    // ---- O += P V  (V^T pre-staged so B-frags are contiguous) -------------
#pragma unroll
    for (int d = 0; d < 8; ++d) {
      BF16x16 bv;
      const __bf16* vp = vtb + (d * 16 + lrow) * 2048 + jb + koff;
      bv.h[0] = *(const v8bf*)vp;
      bv.h[1] = *(const v8bf*)(vp + 16);
      o[d] = __builtin_amdgcn_wmma_f32_16x16x32_bf16(false, ap.v, false, bv.v,
                                                     (short)0, o[d], false, false);
    }
  }

  // ---- normalize + store bf16 [b][n][h*128 + dh] ---------------------------
#pragma unroll
  for (int s = 0; s < 8; ++s) lsum[s] = 1.f / lsum[s];
#pragma unroll
  for (int d = 0; d < 8; ++d)
#pragma unroll
    for (int s = 0; s < 8; ++s) {
      int row = qtile + s + hi * 8;
      att[(b * 2048 + row) * 2048 + h * 128 + d * 16 + lrow] =
          f2bf(o[d][s] * lsum[s]);
    }
}

// ------------------------------- launcher ----------------------------------
extern "C" void kernel_launch(void* const* d_in, const int* in_sizes, int n_in,
                              void* d_out, int out_size, void* d_ws, size_t ws_size,
                              hipStream_t stream) {
  (void)in_sizes; (void)n_in; (void)out_size; (void)ws_size;
  const float* x  = (const float*)d_in[0];
  const float* Wq = (const float*)d_in[1];
  const float* Wk = (const float*)d_in[2];
  const float* Wv = (const float*)d_in[3];
  const float* Wo = (const float*)d_in[4];
  float* out = (float*)d_out;

  char* ws = (char*)d_ws;
  // workspace layout (bytes); att aliases xb (dead after first GEMM)
  __bf16* xb    = (__bf16*)(ws);                          // 8192*2048*2 = 33554432
  __bf16* wqkvT = (__bf16*)(ws + 33554432u);              // 3072*2048*2 = 12582912
  __bf16* wkT   = (__bf16*)(ws + 33554432u + 2048u*2048u*2u);
  __bf16* wvT   = (__bf16*)(ws + 33554432u + 2560u*2048u*2u);
  __bf16* woT   = (__bf16*)(ws + 46137344u);              // 2048*2048*2 = 8388608
  __bf16* qkv   = (__bf16*)(ws + 54525952u);              // 8192*3072*2 = 50331648
  __bf16* vt    = (__bf16*)(ws + 104857600u);             // 2048*2048*2 = 8388608
  __bf16* att   = xb;                                     // alias

  // 1) precision conversion + weight pre-transposes (bf16, [Nout][K])
  cvt_f32_bf16<<<16384, 256, 0, stream>>>(x, xb);
  transpose_cvt<<<16384, 256, 0, stream>>>(Wq, wqkvT, 2048, 2048);
  transpose_cvt<<<4096,  256, 0, stream>>>(Wk, wkT,   2048, 512);
  transpose_cvt<<<4096,  256, 0, stream>>>(Wv, wvT,   2048, 512);
  transpose_cvt<<<16384, 256, 0, stream>>>(Wo, woT,   2048, 2048);

  // 2) fused QKV projection: [8192,2048] x [2048,3072] -> bf16 [8192,3072]
  gemm_bf16<<<dim3(3072 / 64, 8192 / 128), 256, 0, stream>>>(
      xb, wqkvT, qkv, 8192, 3072, 2048, 0);

  // 3) V -> V^T per (b,kvh) so P*V B-fragments are contiguous
  transpose_v<<<16384, 256, 0, stream>>>(qkv, vt);

  // 4) causal GQA flash attention -> bf16 [8192,2048]
  flash_attn<<<dim3(2048 / 64, 64), 128, 0, stream>>>(qkv, vt, att);

  // 5) output projection -> fp32 d_out
  gemm_bf16<<<dim3(2048 / 64, 8192 / 128), 256, 0, stream>>>(
      att, woT, out, 8192, 2048, 2048, 1);
}
